// SimpleLSTM_91070486544429
// MI455X (gfx1250) — compile-verified, matrix operand order checked by reading, not run on hardware
//
#include <hip/hip_runtime.h>
#include <hip/hip_bf16.h>

// ---------------------------------------------------------------------------
// LSTM: B=64, T=512, F=1024, H=1024  (flax gate order i,f,g,o)
//   phase 1: xg = x @ Wx + b     (bf16 WMMA GEMM, f32 accum)
//   phase 2: 512 sequential steps: z = xg[:,t,:] + h @ Wh ; gates ; c,h update
// ---------------------------------------------------------------------------

typedef float  v8f  __attribute__((ext_vector_type(8)));
typedef __bf16 v8bf __attribute__((ext_vector_type(8)));
typedef __bf16 v16bf __attribute__((ext_vector_type(16)));

#define LSTM_B 64
#define LSTM_T 512
#define LSTM_F 1024
#define LSTM_H 1024
#define LSTM_G 4096   // 4*H
#define ROWS  32768   // B*T

__device__ __forceinline__ unsigned short f2bf(float f) {
    unsigned int u = __float_as_uint(f);
    unsigned int r = ((u >> 16) & 1u) + 0x7FFFu;   // round to nearest even
    return (unsigned short)((u + r) >> 16);
}

__device__ __forceinline__ v16bf load_frag(const __bf16* __restrict__ base) {
    // base points at (row-of-fragment)*K + k0 + half8 ; 16B aligned
    v8bf c0 = *reinterpret_cast<const v8bf*>(base);
    v8bf c1 = *reinterpret_cast<const v8bf*>(base + 16);
    return __builtin_shufflevector(c0, c1, 0,1,2,3,4,5,6,7,8,9,10,11,12,13,14,15);
}

__device__ __forceinline__ float sigmoidf_(float x) {
    return 1.0f / (1.0f + __expf(-x));
}

// --------------------------- converters ------------------------------------

__global__ __launch_bounds__(256)
void k_f32_to_bf16(const float* __restrict__ src, unsigned short* __restrict__ dst, int n) {
    int i = (blockIdx.x * 256 + threadIdx.x) * 4;
    if (i + 3 < n) {
        float4 v = *reinterpret_cast<const float4*>(src + i);
        unsigned int lo = (unsigned int)f2bf(v.x) | ((unsigned int)f2bf(v.y) << 16);
        unsigned int hi = (unsigned int)f2bf(v.z) | ((unsigned int)f2bf(v.w) << 16);
        uint2 packed; packed.x = lo; packed.y = hi;
        *reinterpret_cast<uint2*>(dst + i) = packed;
    }
}

// W [K=1024, N=4096] f32 (row-major)  ->  WT [N=4096, K=1024] bf16 (row-major)
__global__ __launch_bounds__(256)
void k_transpose_to_bf16(const float* __restrict__ W, unsigned short* __restrict__ WT) {
    int idx = blockIdx.x * 256 + threadIdx.x;      // over 4096*1024
    int n = idx >> 10;
    int k = idx & 1023;
    WT[idx] = f2bf(W[(size_t)k * LSTM_G + n]);
}

__global__ __launch_bounds__(256)
void k_init_state(float* __restrict__ c, unsigned short* __restrict__ h0, int n) {
    int i = blockIdx.x * 256 + threadIdx.x;
    if (i < n) { c[i] = 0.0f; h0[i] = 0; }
}

// --------------------------- phase 1: xg = x @ Wx + b ----------------------
// Block tile 128(M) x 256(N); 8 waves, each wave owns 32x128 (2x8 WMMA tiles):
// 16 WMMAs per 20 b128 loads per K-step, accumulators stay resident (128 VGPRs).
__global__ __launch_bounds__(256)
void k_gemm_xg(const __bf16* __restrict__ A,    // [32768, 1024] bf16
               const __bf16* __restrict__ BT,   // [4096, 1024]  bf16 (WxT)
               const float*  __restrict__ bias, // [4096]
               float*        __restrict__ C) {  // [32768, 4096]
    const int lane  = threadIdx.x & 31;
    const int wave  = threadIdx.x >> 5;
    const int half8 = (lane & 16) ? 8 : 0;
    const int l15   = lane & 15;
    const int wm    = wave & 3;          // 4 M positions of 32 rows
    const int wn    = wave >> 2;         // 2 N positions of 128 cols
    const int mbase = blockIdx.y * 128 + wm * 32;
    const int nbase = blockIdx.x * 256 + wn * 128;

    v8f acc[2][8] = {};

    for (int k0 = 0; k0 < LSTM_F; k0 += 32) {
        v16bf afrag[2], bfrag[8];
#pragma unroll
        for (int mi = 0; mi < 2; ++mi) {
            const __bf16* p = A + (size_t)(mbase + mi * 16 + l15) * LSTM_F + k0 + half8;
            afrag[mi] = load_frag(p);
            __builtin_prefetch((const void*)(p + 32), 0, 3);  // next K step, WGP scope
        }
#pragma unroll
        for (int ni = 0; ni < 8; ++ni) {
            const __bf16* p = BT + (size_t)(nbase + ni * 16 + l15) * LSTM_F + k0 + half8;
            bfrag[ni] = load_frag(p);
        }
#pragma unroll
        for (int mi = 0; mi < 2; ++mi)
#pragma unroll
            for (int ni = 0; ni < 8; ++ni)
                acc[mi][ni] = __builtin_amdgcn_wmma_f32_16x16x32_bf16(
                    false, afrag[mi], false, bfrag[ni],
                    (short)0, acc[mi][ni], false, false);
    }

#pragma unroll
    for (int mi = 0; mi < 2; ++mi)
#pragma unroll
        for (int ni = 0; ni < 8; ++ni)
#pragma unroll
            for (int r = 0; r < 8; ++r) {
                int row = mbase + mi * 16 + r + half8;
                int col = nbase + ni * 16 + l15;
                C[(size_t)row * LSTM_G + col] = acc[mi][ni][r] + bias[col];
            }
}

// --------------------------- phase 2: one timestep -------------------------
// grid = 64 workgroups (j = 16-col slab of h), 8 waves:
//   wave -> (m = batch tile 0..3, gate pair {0,1} or {2,3})
__global__ __launch_bounds__(256)
void k_step(const float*  __restrict__ xg,    // [32768, 4096]
            const __bf16* __restrict__ WhT,   // [4096, 1024] bf16
            const __bf16* __restrict__ hprev, // [64, 1024]  bf16
            unsigned short* __restrict__ hnext, // [64, 1024] bf16
            float* __restrict__ cstate,       // [64, 1024]
            float* __restrict__ out,          // [64, 512, 1024]
            int t) {
    __shared__ float zbuf[4][4][16][17];      // [m][gate][row][col], padded

    const int lane  = threadIdx.x & 31;
    const int wave  = threadIdx.x >> 5;
    const int half8 = (lane & 16) ? 8 : 0;
    const int l15   = lane & 15;
    const int m     = wave >> 1;        // batch tile
    const int g0    = (wave & 1) * 2;   // gates g0, g0+1
    const int j     = blockIdx.x;       // h-column tile

    v8f acc0 = {}, acc1 = {};

    for (int k0 = 0; k0 < LSTM_H; k0 += 32) {
        const __bf16* pa = hprev + (size_t)(m * 16 + l15) * LSTM_H + k0 + half8;
        v16bf afrag = load_frag(pa);
        const __bf16* pb0 = WhT + (size_t)(g0 * LSTM_H + j * 16 + l15) * LSTM_H + k0 + half8;
        const __bf16* pb1 = WhT + (size_t)((g0 + 1) * LSTM_H + j * 16 + l15) * LSTM_H + k0 + half8;
        v16bf bfrag0 = load_frag(pb0);
        v16bf bfrag1 = load_frag(pb1);
        __builtin_prefetch((const void*)(pb0 + 32), 0, 3);
        acc0 = __builtin_amdgcn_wmma_f32_16x16x32_bf16(false, afrag, false, bfrag0,
                                                       (short)0, acc0, false, false);
        acc1 = __builtin_amdgcn_wmma_f32_16x16x32_bf16(false, afrag, false, bfrag1,
                                                       (short)0, acc1, false, false);
    }

    // z = h@Wh + xg[:,t,:]  (bias already folded into xg); stash to LDS
#pragma unroll
    for (int r = 0; r < 8; ++r) {
        int b = m * 16 + r + half8;                         // batch row
        size_t base = ((size_t)b * LSTM_T + t) * LSTM_G + j * 16 + l15;
        zbuf[m][g0    ][r + half8][l15] = acc0[r] + xg[base + (size_t)g0 * LSTM_H];
        zbuf[m][g0 + 1][r + half8][l15] = acc1[r] + xg[base + (size_t)(g0 + 1) * LSTM_H];
    }
    __syncthreads();

    // gate math: 64 rows x 16 cols = 1024 elems, 256 threads
    for (int idx = threadIdx.x; idx < LSTM_B * 16; idx += 256) {
        int row = idx >> 4;          // batch 0..63
        int col = idx & 15;
        float iz = zbuf[row >> 4][0][row & 15][col];
        float fz = zbuf[row >> 4][1][row & 15][col];
        float gz = zbuf[row >> 4][2][row & 15][col];
        float oz = zbuf[row >> 4][3][row & 15][col];
        int hcol = j * 16 + col;
        float c  = cstate[row * LSTM_H + hcol];
        float nc = sigmoidf_(fz) * c + sigmoidf_(iz) * tanhf(gz);
        float nh = sigmoidf_(oz) * tanhf(nc);
        cstate[row * LSTM_H + hcol] = nc;
        hnext[row * LSTM_H + hcol]  = f2bf(nh);
        out[((size_t)row * LSTM_T + t) * LSTM_H + hcol] = nh;
    }
}

// --------------------------- launcher --------------------------------------

extern "C" void kernel_launch(void* const* d_in, const int* in_sizes, int n_in,
                              void* d_out, int out_size, void* d_ws, size_t ws_size,
                              hipStream_t stream) {
    const float* x    = (const float*)d_in[0];   // [64,512,1024]
    const float* Wx   = (const float*)d_in[1];   // [1024,4096]
    const float* Wh   = (const float*)d_in[2];   // [1024,4096]
    const float* bias = (const float*)d_in[3];   // [4096]
    float* out = (float*)d_out;                  // [64,512,1024]

    char* ws = (char*)d_ws;
    const size_t OFF_XG  = 0;                                   // 32768*4096*4
    const size_t OFF_XBF = OFF_XG  + (size_t)ROWS * LSTM_G * 4; // 32768*1024*2
    const size_t OFF_WXT = OFF_XBF + (size_t)ROWS * LSTM_F * 2; // 4096*1024*2
    const size_t OFF_WHT = OFF_WXT + (size_t)LSTM_G * LSTM_F * 2;
    const size_t OFF_C   = OFF_WHT + (size_t)LSTM_G * LSTM_H * 2;
    const size_t OFF_H0  = OFF_C   + (size_t)LSTM_B * LSTM_H * 4;
    const size_t OFF_H1  = OFF_H0  + (size_t)LSTM_B * LSTM_H * 2;

    float*          xg  = (float*)(ws + OFF_XG);
    unsigned short* xbf = (unsigned short*)(ws + OFF_XBF);
    unsigned short* wxt = (unsigned short*)(ws + OFF_WXT);
    unsigned short* wht = (unsigned short*)(ws + OFF_WHT);
    float*          cst = (float*)(ws + OFF_C);
    unsigned short* h0  = (unsigned short*)(ws + OFF_H0);
    unsigned short* h1  = (unsigned short*)(ws + OFF_H1);

    // 1) convert / transpose to bf16
    const int NX = ROWS * LSTM_F;                  // 33554432
    k_f32_to_bf16<<<NX / 1024, 256, 0, stream>>>(x, xbf, NX);
    k_transpose_to_bf16<<<(LSTM_G * LSTM_F) / 256, 256, 0, stream>>>(Wx, wxt);
    k_transpose_to_bf16<<<(LSTM_G * LSTM_H) / 256, 256, 0, stream>>>(Wh, wht);
    k_init_state<<<(LSTM_B * LSTM_H) / 256, 256, 0, stream>>>(cst, h0, LSTM_B * LSTM_H);

    // 2) big input-projection GEMM (bias folded in)
    dim3 ggrid(LSTM_G / 256, ROWS / 128);          // (16, 256)
    k_gemm_xg<<<ggrid, 256, 0, stream>>>((const __bf16*)xbf, (const __bf16*)wxt, bias, xg);

    // 3) sequential recurrence, ping-pong h buffers
    for (int t = 0; t < LSTM_T; ++t) {
        unsigned short* hp = (t & 1) ? h1 : h0;
        unsigned short* hn = (t & 1) ? h0 : h1;
        k_step<<<LSTM_H / 16, 256, 0, stream>>>(xg, (const __bf16*)wht,
                                                (const __bf16*)hp, hn, cst, out, t);
    }
    (void)in_sizes; (void)n_in; (void)out_size; (void)ws_size;
}
